// AxialAttention_44092134260808
// MI455X (gfx1250) — matrix operand order repaired and checked by reading, not compiled, not run on hardware
//
#include <hip/hip_runtime.h>

// ---------------------------------------------------------------------------
// Music-Transformer relative attention for MI455X (gfx1250, wave32, WMMA).
//   B=4, N=1024, C=1024, H=16, DH=64.
// All GEMMs via v_wmma_f32_16x16x32_f16. Attention is a single-pass flash
// kernel: scores, relative bands and P@V' never touch HBM. Tile staging goes
// through GLOBAL_LOAD_ASYNC_TO_LDS (ASYNCcnt) when the toolchain exposes it.
// ---------------------------------------------------------------------------

typedef __attribute__((ext_vector_type(16))) _Float16 v16h;
typedef __attribute__((ext_vector_type(8)))  float    v8f;
typedef int v4i __attribute__((vector_size(16)));

constexpr int NB    = 4;
constexpr int NSEQ  = 1024;
constexpr int CDIM  = 1024;
constexpr int NH    = 16;
constexpr int DHEAD = 64;

#if defined(__has_builtin)
#if __has_builtin(__builtin_amdgcn_global_load_async_to_lds_b128)
#define ASYNC_COPY 1
#endif
#endif
#ifndef ASYNC_COPY
#define ASYNC_COPY 0
#endif

// 16-byte global->LDS copy, async (ASYNCcnt) when available.
// Builtin signature (from probe): (v4i addrspace(1)*, v4i addrspace(3)*,
// imm offset, imm cpol).
__device__ __forceinline__ void cp16(_Float16* l, const _Float16* g) {
#if ASYNC_COPY
  __builtin_amdgcn_global_load_async_to_lds_b128(
      (__attribute__((address_space(1))) v4i*)g,
      (__attribute__((address_space(3))) v4i*)l, 0, 0);
#else
  *(uint4*)l = *(const uint4*)g;
#endif
}

__device__ __forceinline__ void async_wait() {
#if ASYNC_COPY
  asm volatile("s_wait_asynccnt 0x0" ::: "memory");
#endif
}

// ---------------- WMMA helpers (layouts per CDNA5 ISA) ----------

__device__ __forceinline__ v8f wmma_f16(v16h a, v16h b, v8f c) {
  return __builtin_amdgcn_wmma_f32_16x16x32_f16(
      /*neg_a=*/false, a, /*neg_b=*/false, b,
      /*c_mod=*/(short)0, c, /*reuse_a=*/false, /*reuse_b=*/false);
}

// A fragment 16x32 f16 from row-major LDS [row][k]. lane = M, half-wave picks
// the K interleave (K 0..7 / 16..23, +8 for lanes 16-31). Two contiguous
// 8-half runs -> compiler emits 2x ds_load_b128.
__device__ __forceinline__ v16h load_A(const _Float16* s, int m0, int k0, int ld) {
  const int lane = threadIdx.x & 31;
  const int m = lane & 15;
  const int h = lane >> 4;
  const _Float16* p = s + (size_t)(m0 + m) * ld + k0;
  v16h a;
#pragma unroll
  for (int e = 0; e < 16; ++e) {
    const int r = e >> 1;
    const int k = (r < 4) ? (2 * r + (e & 1) + 8 * h)
                          : (16 + 2 * (r - 4) + (e & 1) + 8 * h);
    a[e] = p[k];
  }
  return a;
}

// B fragment 32x16 where B[k][n] = Src[n][k] (A @ Src^T). lane = N,
// element e -> K = k0 + 8*(lane>=16) + (e < 8 ? e : e + 8). Two 8-half runs.
__device__ __forceinline__ v16h load_Bt(const _Float16* s, int n0, int k0, int ld) {
  const int lane = threadIdx.x & 31;
  const int n = lane & 15;
  const int h = lane >> 4;
  const _Float16* p = s + (size_t)(n0 + n) * ld + k0 + 8 * h;
  v16h b;
#pragma unroll
  for (int e = 0; e < 16; ++e) b[e] = p[e < 8 ? e : e + 8];
  return b;
}

// D fragment store: element r -> row m0 + r + 8*(lane>=16), col n0 + (lane&15).
__device__ __forceinline__ void store_D(float* s, v8f d, int m0, int n0, int ld) {
  const int lane = threadIdx.x & 31;
  const int col = lane & 15;
  const int h = lane >> 4;
  float* p = s + (size_t)(m0 + 8 * h) * ld + n0 + col;
#pragma unroll
  for (int r = 0; r < 8; ++r) p[(size_t)r * ld] = d[r];
}

// ---------------- fp32 -> f16 conversion -----------------------------------

__global__ void cvt_f32_f16(const float* __restrict__ in,
                            _Float16* __restrict__ out, int n) {
  int i = blockIdx.x * blockDim.x + threadIdx.x;
  if (i < n) out[i] = (_Float16)in[i];
}

// ---------------- GEMM: C[M,Nn] = A[M,K] @ W^T (+bias) ---------------------
// A row-major f16 [M,K]; W row-major f16 [Nn,K]. 256 thr = 8 waves.
// Block tile 128x64, wave w owns rows 16w..16w+15, 4 col WMMA tiles.

template <bool F32OUT>
__global__ __launch_bounds__(256) void gemm_wt(
    const _Float16* __restrict__ A, const _Float16* __restrict__ W,
    const float* __restrict__ bias, void* __restrict__ Cout,
    int M, int Nn, int K) {
  __shared__ _Float16 As[128 * 48];
  __shared__ _Float16 Ws[64 * 48];

  const int tid = threadIdx.x;
  const int lane = tid & 31;
  const int w = tid >> 5;
  const int row0 = blockIdx.y * 128;
  const int n0g = blockIdx.x * 64;

  v8f acc[4];
  v8f zero = {};
#pragma unroll
  for (int i = 0; i < 4; ++i) acc[i] = zero;

  for (int kk = 0; kk < K; kk += 32) {
    __syncthreads();
    // As: 128 rows x 32 halves = 512 x 16B; 2 chunks per thread.
#pragma unroll
    for (int c = tid; c < 512; c += 256) {
      const int r = c >> 2, ch = c & 3;
      cp16(&As[r * 48 + ch * 8], &A[(size_t)(row0 + r) * K + kk + ch * 8]);
    }
    // Ws: 64 rows x 32 halves = 256 x 16B; 1 per thread.
    {
      const int r = tid >> 2, ch = tid & 3;
      cp16(&Ws[r * 48 + ch * 8], &W[(size_t)(n0g + r) * K + kk + ch * 8]);
    }
    async_wait();
    __syncthreads();

    const v16h a = load_A(As, 16 * w, 0, 48);
#pragma unroll
    for (int ct = 0; ct < 4; ++ct) {
      const v16h b = load_Bt(Ws, ct * 16, 0, 48);
      acc[ct] = wmma_f16(a, b, acc[ct]);
    }
  }

  const int col = lane & 15;
  const int hh = lane >> 4;
#pragma unroll
  for (int ct = 0; ct < 4; ++ct) {
#pragma unroll
    for (int r = 0; r < 8; ++r) {
      const int row = row0 + 16 * w + r + 8 * hh;
      const int nn = n0g + ct * 16 + col;
      const float v = acc[ct][r];
      if (F32OUT)
        ((float*)Cout)[(size_t)row * Nn + nn] = v + bias[nn];
      else
        ((_Float16*)Cout)[(size_t)row * Nn + nn] = (_Float16)v;
    }
  }
}

// ---------------- Flash attention with skewed relative positions -----------
// grid: (N/64, B*H), block: 128 thr = 4 waves. Wave w owns query rows
// 16w..16w+15 of the 64-row block. Per 64-col KV tile:
//   S  = Q Kc^T                              (WMMA)
//   Erel[i, t] = q[i]·ErQ[dbase+t] + k[i]·ErK[dbase+t], t in [0,128)  (WMMA)
//   score[i,j] = (S + Erel[i, j-i+63]) * DH^-1/2   (skew gather)
//   online softmax; O += P @ (V + Er_v[j])   (WMMA, V kept transposed in LDS)

__global__ __launch_bounds__(128) void attn_rel(
    const _Float16* __restrict__ qg, const _Float16* __restrict__ kg,
    const _Float16* __restrict__ vg, const _Float16* __restrict__ erq,
    const _Float16* __restrict__ erk, const _Float16* __restrict__ erv,
    _Float16* __restrict__ og) {
  extern __shared__ char smem[];
  _Float16* Qs  = (_Float16*)smem;          // 64x64
  _Float16* Krs = Qs + 64 * 64;             // K rows at query indices
  _Float16* Ks  = Krs + 64 * 64;            // K cols (current tile)
  _Float16* VsT = Ks + 64 * 64;             // (V+Er_v)^T : [64 dims][64 rows], ld 72
  _Float16* EqS = VsT + 64 * 72;            // 128x64 Er_q slice
  _Float16* EkS = EqS + 128 * 64;           // 128x64 Er_k slice
  float* Erel = (float*)(EkS + 128 * 64);   // 64x128
  float* Ss   = Erel + 64 * 128;            // 64x64 scores
  _Float16* Ps = (_Float16*)(Ss + 64 * 64); // 64x64 probs f16
  float* mrow = (float*)(Ps + 64 * 64);     // 64
  float* lrow = mrow + 64;                  // 64

  const int tid = threadIdx.x;
  const int lane = tid & 31;
  const int w = tid >> 5;
  const int m0 = 16 * w;
  const int hh = lane >> 4;
  const int b = blockIdx.y / NH;
  const int h = blockIdx.y % NH;
  const int i0 = blockIdx.x * 64;
  const float scale = 0.125f;  // 64^-0.5

  const _Float16* qbase = qg + (size_t)b * NSEQ * CDIM + h * DHEAD;
  const _Float16* kbase = kg + (size_t)b * NSEQ * CDIM + h * DHEAD;
  const _Float16* vbase = vg + (size_t)b * NSEQ * CDIM + h * DHEAD;

  // Q rows and K rows (query-indexed, for the KEr term): async into LDS.
  for (int c = tid; c < 64 * 8; c += 128) {
    const int r = c >> 3, ch = c & 7;
    cp16(&Qs[r * 64 + ch * 8], &qbase[(size_t)(i0 + r) * CDIM + ch * 8]);
    cp16(&Krs[r * 64 + ch * 8], &kbase[(size_t)(i0 + r) * CDIM + ch * 8]);
  }
  for (int r = tid; r < 64; r += 128) { mrow[r] = -1e30f; lrow[r] = 0.0f; }

  v8f acc[4];
  v8f zero = {};
#pragma unroll
  for (int i = 0; i < 4; ++i) acc[i] = zero;

  for (int j0 = 0; j0 < NSEQ; j0 += 64) {
    __syncthreads();
    // K column tile (async).
    for (int c = tid; c < 64 * 8; c += 128) {
      const int r = c >> 3, ch = c & 7;
      cp16(&Ks[r * 64 + ch * 8], &kbase[(size_t)(j0 + r) * CDIM + ch * 8]);
    }
    // Er slices: d = dbase + t for t in [0,128); zero-fill out of range.
    const int dbase = (j0 - i0) + (NSEQ - 1) - 63;
    for (int c = tid; c < 128 * 8; c += 128) {
      const int t = c >> 3, ch = c & 7;
      const int d = dbase + t;
      if (d >= 0 && d < NSEQ) {
        cp16(&EqS[t * 64 + ch * 8], &erq[(size_t)d * DHEAD + ch * 8]);
        cp16(&EkS[t * 64 + ch * 8], &erk[(size_t)d * DHEAD + ch * 8]);
      } else {
        const uint4 z = {0, 0, 0, 0};
        *(uint4*)&EqS[t * 64 + ch * 8] = z;
        *(uint4*)&EkS[t * 64 + ch * 8] = z;
      }
    }
    // V' = V + Er_v[j], stored transposed so P@V' uses contiguous B loads.
    for (int c = tid; c < 64 * 64; c += 128) {
      const int r = c >> 6, d = c & 63;  // consecutive lanes: consecutive d
      const float vv = (float)vbase[(size_t)(j0 + r) * CDIM + d] +
                       (float)erv[(size_t)(j0 + r) * DHEAD + d];
      VsT[d * 72 + r] = (_Float16)vv;
    }
    async_wait();
    __syncthreads();

    // S = Q Kc^T  (4 col tiles x 2 k-steps per wave).
#pragma unroll
    for (int ct = 0; ct < 4; ++ct) {
      v8f s = zero;
#pragma unroll
      for (int kk = 0; kk < 64; kk += 32) {
        const v16h a = load_A(Qs, m0, kk, 64);
        const v16h bq = load_Bt(Ks, ct * 16, kk, 64);
        s = wmma_f16(a, bq, s);
      }
      store_D(Ss, s, m0, ct * 16, 64);
    }
    // Erel = Q ErQ^T + Krow ErK^T  (8 col tiles over 128 x 2 k-steps x 2).
#pragma unroll
    for (int ct = 0; ct < 8; ++ct) {
      v8f eq = zero, ek = zero;
#pragma unroll
      for (int kk = 0; kk < 64; kk += 32) {
        const v16h aq = load_A(Qs, m0, kk, 64);
        const v16h ak = load_A(Krs, m0, kk, 64);
        const v16h bq = load_Bt(EqS, ct * 16, kk, 64);
        const v16h bk = load_Bt(EkS, ct * 16, kk, 64);
        eq = wmma_f16(aq, bq, eq);
        ek = wmma_f16(ak, bk, ek);
      }
      store_D(Erel, eq + ek, m0, ct * 16, 128);
    }

    // Online softmax over this wave's 16 rows; lane pair (l, l^16) splits the
    // 64 columns. Pass 1 folds the skew gather into Ss so pass 2 reads once.
    const int rowl = m0 + (lane & 15);
    float tmax = -1e30f;
    for (int jc = 0; jc < 32; ++jc) {
      const int jl = hh * 32 + jc;
      const float sc =
          (Ss[rowl * 64 + jl] + Erel[rowl * 128 + (jl - rowl + 63)]) * scale;
      Ss[rowl * 64 + jl] = sc;
      tmax = fmaxf(tmax, sc);
    }
    tmax = fmaxf(tmax, __shfl_xor(tmax, 16));
    const float mold = mrow[rowl];
    const float mnew = fmaxf(mold, tmax);
    const float fct = __expf(mold - mnew);
    float lsum = 0.0f;
    for (int jc = 0; jc < 32; ++jc) {
      const int jl = hh * 32 + jc;
      const float p = __expf(Ss[rowl * 64 + jl] - mnew);
      Ps[rowl * 64 + jl] = (_Float16)p;
      lsum += p;
    }
    lsum += __shfl_xor(lsum, 16);
    if (hh == 0) {
      mrow[rowl] = mnew;
      lrow[rowl] = lrow[rowl] * fct + lsum;
    }

    // Rescale accumulators and add P @ V' (B from transposed V: load_Bt).
#pragma unroll
    for (int ct = 0; ct < 4; ++ct) {
#pragma unroll
      for (int r = 0; r < 8; ++r) {
        const float fr = __shfl(fct, r + 8 * hh);
        acc[ct][r] *= fr;
      }
    }
#pragma unroll
    for (int kk = 0; kk < 64; kk += 32) {
      const v16h a = load_A(Ps, m0, kk, 64);
#pragma unroll
      for (int ct = 0; ct < 4; ++ct) {
        const v16h bv = load_Bt(VsT, ct * 16, kk, 72);
        acc[ct] = wmma_f16(a, bv, acc[ct]);
      }
    }
  }

  // Normalize and store O (f16, [B,N,C] layout for the output projection).
  const int col = lane & 15;
#pragma unroll
  for (int ct = 0; ct < 4; ++ct) {
#pragma unroll
    for (int r = 0; r < 8; ++r) {
      const int re = m0 + r + 8 * hh;
      const float o = acc[ct][r] / lrow[re];
      og[((size_t)b * NSEQ + i0 + re) * CDIM + h * DHEAD + ct * 16 + col] =
          (_Float16)o;
    }
  }
}

// ---------------------------------------------------------------------------

extern "C" void kernel_launch(void* const* d_in, const int* in_sizes, int n_in,
                              void* d_out, int out_size, void* d_ws,
                              size_t ws_size, hipStream_t stream) {
  (void)n_in; (void)out_size; (void)ws_size;
  const float* x   = (const float*)d_in[0];
  const float* Wq  = (const float*)d_in[1];
  const float* Wk  = (const float*)d_in[2];
  const float* Wv  = (const float*)d_in[3];
  const float* Erq = (const float*)d_in[4];
  const float* Erk = (const float*)d_in[5];
  const float* Erv = (const float*)d_in[6];
  const float* Wp  = (const float*)d_in[7];
  const float* bp  = (const float*)d_in[8];

  const int nX = NB * NSEQ * CDIM;      // 4,194,304
  const int nW = CDIM * CDIM;           // 1,048,576
  const int nE = in_sizes[4];           // MAX_LEN * DH

  char* p = (char*)d_ws;
  auto carve = [&](size_t bytes) {
    char* r = p;
    p += (bytes + 255) & ~(size_t)255;
    return r;
  };
  _Float16* xh  = (_Float16*)carve((size_t)nX * 2);
  _Float16* wqh = (_Float16*)carve((size_t)nW * 2);
  _Float16* wkh = (_Float16*)carve((size_t)nW * 2);
  _Float16* wvh = (_Float16*)carve((size_t)nW * 2);
  _Float16* wph = (_Float16*)carve((size_t)nW * 2);
  _Float16* eqh = (_Float16*)carve((size_t)nE * 2);
  _Float16* ekh = (_Float16*)carve((size_t)nE * 2);
  _Float16* evh = (_Float16*)carve((size_t)nE * 2);
  _Float16* qh  = (_Float16*)carve((size_t)nX * 2);
  _Float16* kh  = (_Float16*)carve((size_t)nX * 2);
  _Float16* vh  = (_Float16*)carve((size_t)nX * 2);
  _Float16* oh  = (_Float16*)carve((size_t)nX * 2);

  auto cvt = [&](const float* in, _Float16* out, int n) {
    cvt_f32_f16<<<(n + 255) / 256, 256, 0, stream>>>(in, out, n);
  };
  cvt(x, xh, nX);
  cvt(Wq, wqh, nW);
  cvt(Wk, wkh, nW);
  cvt(Wv, wvh, nW);
  cvt(Wp, wph, nW);
  cvt(Erq, eqh, nE);
  cvt(Erk, ekh, nE);
  cvt(Erv, evh, nE);

  const int M = NB * NSEQ;  // 4096
  dim3 ggrid(CDIM / 64, M / 128);
  gemm_wt<false><<<ggrid, 256, 0, stream>>>(xh, wqh, nullptr, qh, M, CDIM, CDIM);
  gemm_wt<false><<<ggrid, 256, 0, stream>>>(xh, wkh, nullptr, kh, M, CDIM, CDIM);
  gemm_wt<false><<<ggrid, 256, 0, stream>>>(xh, wvh, nullptr, vh, M, CDIM, CDIM);

  // Dynamic LDS: Q/Kr/Kc (24KB) + VsT (9KB) + Er slices (32KB) + Erel (32KB)
  // + S (16KB) + P (8KB) + row state (512B) = ~121.5KB (<320KB/WGP).
  const size_t att_smem = (size_t)(3 * 64 * 64 + 64 * 72 + 2 * 128 * 64) * 2 +
                          (size_t)(64 * 128 + 64 * 64) * 4 +
                          (size_t)(64 * 64) * 2 + 128 * 4;
  dim3 agrid(NSEQ / 64, NB * NH);
  attn_rel<<<agrid, 128, att_smem, stream>>>(qh, kh, vh, eqh, ekh, evh, oh);

  gemm_wt<true><<<ggrid, 256, 0, stream>>>(oh, wph, bp, (float*)d_out, M, CDIM,
                                           CDIM);
}
